// SelfAttention_16037407883609
// MI455X (gfx1250) — matrix operand order repaired and checked by reading, not compiled, not run on hardware
//
#include <hip/hip_runtime.h>

// ---------------------------------------------------------------------------
// CDNA5 (gfx1250, wave32) self-attention block built entirely on
// v_wmma_f32_16x16x32_f16.  All GEMM-like math (QKV projection, Q*K^T,
// P*V, output projection) goes through the matrix cores with f32 accum.
// Working set (~40MB f16) lives in the 192MB L2; HBM traffic is negligible.
// 32x32 tile per wave in the GEMMs: fits the VGPR budget with zero spills
// (64x64 tiles spill accumulators to scratch).
// ---------------------------------------------------------------------------

typedef _Float16 v16h __attribute__((ext_vector_type(16)));
typedef _Float16 v8h  __attribute__((ext_vector_type(8)));
typedef float    v8f  __attribute__((ext_vector_type(8)));

#define WMMA_F16(a, b, c) \
  __builtin_amdgcn_wmma_f32_16x16x32_f16(false, (a), false, (b), (short)0, (c), false, false)

union AFrag { v16h v; v8h h[2]; };

// Problem constants
#define BATCH   2
#define SEQ     2048
#define EMBD    1024
#define NHEAD   16
#define HDIM    64
#define M_ROWS  (BATCH * SEQ)      // 4096
#define N_QKV   (3 * EMBD)         // 3072

// ------------------------------- prep kernels ------------------------------

__global__ void cvt_f32_to_f16_kernel(const float* __restrict__ src,
                                      _Float16* __restrict__ dst, int n) {
  int i = blockIdx.x * blockDim.x + threadIdx.x;
  if (i < n) dst[i] = (_Float16)src[i];
}

// src is [K][N] row-major f32; dst is [N][K] row-major f16
__global__ void transpose_cvt_kernel(const float* __restrict__ src,
                                     _Float16* __restrict__ dst, int K, int N) {
  int i = blockIdx.x * blockDim.x + threadIdx.x;
  if (i < K * N) {
    int k = i / N, n = i % N;
    dst[(size_t)n * K + k] = (_Float16)src[i];
  }
}

// ------------------------- 32x32-per-wave WMMA GEMM ------------------------
// A: [M][K] f16 row-major.  Bt: [N][K] f16 row-major (i.e. B transposed).
// Each wave accumulates a 32x32 tile as 2x2 WMMA fragments.
// A-fragment lane layout (16x32 f16): lane<16 -> row=lane, k=0..15;
// lane>=16 -> row=lane-16, k=16..31.
// B-fragment lane layout (32x16 f16): lane<16 -> col=lane, k=0..15;
// lane>=16 -> col=lane-16, k=16..31.  With Bt row-major these are single
// contiguous 32B loads per lane.
__device__ __forceinline__ void gemm_tile_32x32(const _Float16* __restrict__ A,
                                                const _Float16* __restrict__ Bt,
                                                int K, int row0, int col0,
                                                v8f c[4]) {
  const int lane = threadIdx.x & 31;
  const int lrow = lane & 15;
  const int lhi  = lane >> 4;
  const _Float16* a0p = A  + (size_t)(row0 + lrow) * K + lhi * 16;
  const _Float16* a1p = a0p + (size_t)16 * K;
  const _Float16* b0p = Bt + (size_t)(col0 + lrow) * K + lhi * 16;
  const _Float16* b1p = b0p + (size_t)16 * K;
  for (int kb = 0; kb < K; kb += 32) {
    AFrag a0, a1;
    a0.h[0] = *(const v8h*)(a0p + kb);
    a0.h[1] = *(const v8h*)(a0p + kb + 8);
    a1.h[0] = *(const v8h*)(a1p + kb);
    a1.h[1] = *(const v8h*)(a1p + kb + 8);
    v16h b0 = *(const v16h*)(b0p + kb);
    v16h b1 = *(const v16h*)(b1p + kb);
    c[0] = WMMA_F16(a0.v, b0, c[0]);
    c[1] = WMMA_F16(a0.v, b1, c[1]);
    c[2] = WMMA_F16(a1.v, b0, c[2]);
    c[3] = WMMA_F16(a1.v, b1, c[3]);
  }
}

// ------------------------------ QKV projection -----------------------------
// qkv = x @ w_attn + b_attn, routed into:
//   qh [B,H,T,D] f16 (pre-scaled by 1/sqrt(D))
//   kh [B,H,T,D] f16
//   vT [B,H,D,T] f16  (transposed so P*V B-fragments are contiguous)
__global__ void qkv_gemm_kernel(const _Float16* __restrict__ xh,
                                const _Float16* __restrict__ wT,
                                const float* __restrict__ bias,
                                _Float16* __restrict__ qh,
                                _Float16* __restrict__ kh,
                                _Float16* __restrict__ vT) {
  const int NT = N_QKV / 32;                 // 96
  int mi = blockIdx.x / NT, ni = blockIdx.x % NT;
  int row0 = mi * 32, col0 = ni * 32;
  v8f zero = {0.f, 0.f, 0.f, 0.f, 0.f, 0.f, 0.f, 0.f};
  v8f c[4] = {zero, zero, zero, zero};
  gemm_tile_32x32(xh, wT, EMBD, row0, col0, c);

  const int lane = threadIdx.x & 31;
  const int lrow = lane & 15, lhi = lane >> 4;
  #pragma unroll
  for (int i = 0; i < 2; i++) {
    #pragma unroll
    for (int j = 0; j < 2; j++) {
      int col = col0 + j * 16 + lrow;
      float bv = bias[col];
      #pragma unroll
      for (int r = 0; r < 8; r++) {
        int mrow = row0 + i * 16 + r + 8 * lhi;  // C layout: elem r -> rows r / r+8
        float val = c[i * 2 + j][r] + bv;
        int b = mrow >> 11, tt = mrow & (SEQ - 1);
        if (col < EMBD) {
          int h = col >> 6, d = col & 63;
          qh[((size_t)((b * NHEAD + h) * SEQ + tt)) * HDIM + d] =
              (_Float16)(val * 0.125f);     // fold 1/sqrt(64) into Q
        } else if (col < 2 * EMBD) {
          int cc = col - EMBD, h = cc >> 6, d = cc & 63;
          kh[((size_t)((b * NHEAD + h) * SEQ + tt)) * HDIM + d] = (_Float16)val;
        } else {
          int cc = col - 2 * EMBD, h = cc >> 6, d = cc & 63;
          vT[((size_t)((b * NHEAD + h) * HDIM + d)) * SEQ + tt] = (_Float16)val;
        }
      }
    }
  }
}

// ------------------------------ flash attention ----------------------------
// One wave per (b, h, 16-row q-tile).  Online softmax over 32-key blocks.
__global__ void flash_attn_kernel(const _Float16* __restrict__ qh,
                                  const _Float16* __restrict__ kh,
                                  const _Float16* __restrict__ vT,
                                  _Float16* __restrict__ yh) {
  __shared__ __align__(16) _Float16 pbuf[16 * 32];  // P tile staging (A-layout gather)

  int bid = blockIdx.x;
  int qt = bid & (SEQ / 16 - 1);        // 0..127
  int h  = (bid >> 7) & (NHEAD - 1);
  int b  = bid >> 11;
  int q0 = qt * 16;
  const int lane = threadIdx.x & 31, lrow = lane & 15, lhi = lane >> 4;

  size_t headoff = (size_t)(b * NHEAD + h) * SEQ * HDIM;
  const _Float16* qbase = qh + headoff;
  const _Float16* kbase = kh + headoff;
  const _Float16* vbase = vT + headoff;   // [D][T]

  // Q fragments (16x64 tile = two 16x32 A-fragments), loaded once.
  AFrag aq0, aq1;
  const _Float16* qptr = qbase + (size_t)(q0 + lrow) * HDIM + lhi * 16;
  aq0.h[0] = *(const v8h*)(qptr);
  aq0.h[1] = *(const v8h*)(qptr + 8);
  aq1.h[0] = *(const v8h*)(qptr + 32);
  aq1.h[1] = *(const v8h*)(qptr + 40);

  v8f zero = {0.f, 0.f, 0.f, 0.f, 0.f, 0.f, 0.f, 0.f};
  v8f o[4] = {zero, zero, zero, zero};    // 16 rows x 64 dims accumulator
  float mst[8], lst[8];
  #pragma unroll
  for (int r = 0; r < 8; r++) { mst[r] = -1e30f; lst[r] = 0.f; }

  int nkb = (q0 + 16 + 31) >> 5;          // causal: key blocks covering q0+15
  for (int jb = 0; jb < nkb; jb++) {
    int kb0 = jb * 32;

    // S = Q * K^T for 32 keys (two 16-key subtiles, D=64 inner -> 4 WMMAs)
    v8f s0 = zero, s1 = zero;
    {
      const _Float16* kp0 = kbase + (size_t)(kb0 + lrow) * HDIM + lhi * 16;
      const _Float16* kp1 = kp0 + 16 * HDIM;
      v16h bk;
      bk = *(const v16h*)(kp0);      s0 = WMMA_F16(aq0.v, bk, s0);
      bk = *(const v16h*)(kp0 + 32); s0 = WMMA_F16(aq1.v, bk, s0);
      bk = *(const v16h*)(kp1);      s1 = WMMA_F16(aq0.v, bk, s1);
      bk = *(const v16h*)(kp1 + 32); s1 = WMMA_F16(aq1.v, bk, s1);
    }

    // Causal mask + online softmax.  C-layout: elem r <-> row r (lanes 0-15)
    // or r+8 (lanes 16-31); col = lane%16.  Row reductions stay inside each
    // 16-lane half via xor shuffles 1,2,4,8.
    int key0 = kb0 + lrow, key1 = key0 + 16;
    #pragma unroll
    for (int r = 0; r < 8; r++) {
      int qrow = q0 + r + 8 * lhi;
      float x0 = (key0 <= qrow) ? s0[r] : -1e30f;
      float x1 = (key1 <= qrow) ? s1[r] : -1e30f;
      float bm = fmaxf(x0, x1);
      bm = fmaxf(bm, __shfl_xor(bm, 1, 32));
      bm = fmaxf(bm, __shfl_xor(bm, 2, 32));
      bm = fmaxf(bm, __shfl_xor(bm, 4, 32));
      bm = fmaxf(bm, __shfl_xor(bm, 8, 32));
      float mnew  = fmaxf(mst[r], bm);
      float alpha = __expf(mst[r] - mnew);
      float p0 = __expf(x0 - mnew);
      float p1 = __expf(x1 - mnew);
      float rs = p0 + p1;
      rs += __shfl_xor(rs, 1, 32);
      rs += __shfl_xor(rs, 2, 32);
      rs += __shfl_xor(rs, 4, 32);
      rs += __shfl_xor(rs, 8, 32);
      lst[r] = lst[r] * alpha + rs;
      mst[r] = mnew;
      o[0][r] *= alpha; o[1][r] *= alpha; o[2][r] *= alpha; o[3][r] *= alpha;
      // stash P (C-layout) into LDS row-major [16 rows][32 keys]
      pbuf[(r + 8 * lhi) * 32 + lrow]      = (_Float16)p0;
      pbuf[(r + 8 * lhi) * 32 + 16 + lrow] = (_Float16)p1;
    }
    __syncthreads();

    // Regather P in A-fragment layout (contiguous 16B chunks per lane)
    AFrag pa;
    const _Float16* pp = &pbuf[lrow * 32 + lhi * 16];
    pa.h[0] = *(const v8h*)(pp);
    pa.h[1] = *(const v8h*)(pp + 8);

    // O += P * V : V B-fragments contiguous thanks to vT [D][T] layout
    const _Float16* vp = vbase + (size_t)lrow * SEQ + kb0 + lhi * 16;
    #pragma unroll
    for (int t = 0; t < 4; t++) {
      v16h bv = *(const v16h*)(vp + (size_t)(t * 16) * SEQ);
      o[t] = WMMA_F16(pa.v, bv, o[t]);
    }
    __syncthreads();
  }

  // Normalize and write y (f16, [B*T][C] row-major) for the projection GEMM
  float inv[8];
  #pragma unroll
  for (int r = 0; r < 8; r++) inv[r] = 1.0f / lst[r];
  #pragma unroll
  for (int t = 0; t < 4; t++) {
    int col = h * HDIM + t * 16 + lrow;
    #pragma unroll
    for (int r = 0; r < 8; r++) {
      int row = b * SEQ + q0 + r + 8 * lhi;
      yh[(size_t)row * EMBD + col] = (_Float16)(o[t][r] * inv[r]);
    }
  }
}

// ------------------------------ output projection --------------------------
__global__ void proj_gemm_kernel(const _Float16* __restrict__ yh,
                                 const _Float16* __restrict__ wT,
                                 const float* __restrict__ bias,
                                 float* __restrict__ out) {
  const int NT = EMBD / 32;                  // 32
  int mi = blockIdx.x / NT, ni = blockIdx.x % NT;
  int row0 = mi * 32, col0 = ni * 32;
  v8f zero = {0.f, 0.f, 0.f, 0.f, 0.f, 0.f, 0.f, 0.f};
  v8f c[4] = {zero, zero, zero, zero};
  gemm_tile_32x32(yh, wT, EMBD, row0, col0, c);

  const int lane = threadIdx.x & 31;
  const int lrow = lane & 15, lhi = lane >> 4;
  #pragma unroll
  for (int i = 0; i < 2; i++) {
    #pragma unroll
    for (int j = 0; j < 2; j++) {
      int col = col0 + j * 16 + lrow;
      float bv = bias[col];
      #pragma unroll
      for (int r = 0; r < 8; r++) {
        int mrow = row0 + i * 16 + r + 8 * lhi;
        out[(size_t)mrow * EMBD + col] = c[i * 2 + j][r] + bv;
      }
    }
  }
}

// --------------------------------- launcher --------------------------------
extern "C" void kernel_launch(void* const* d_in, const int* in_sizes, int n_in,
                              void* d_out, int out_size, void* d_ws, size_t ws_size,
                              hipStream_t stream) {
  const float* x      = (const float*)d_in[0];  // [B,T,C]
  const float* w_attn = (const float*)d_in[1];  // [C, 3C]
  const float* b_attn = (const float*)d_in[2];  // [3C]
  const float* w_proj = (const float*)d_in[3];  // [C, C]
  const float* b_proj = (const float*)d_in[4];  // [C]
  float* out = (float*)d_out;                   // [B,T,C] f32

  char* ws = (char*)d_ws;
  // f16 workspace layout (yh reuses xh's slot: xh dead after QKV GEMM)
  _Float16* xh     = (_Float16*)(ws);                     //  8 MB [4096][1024]
  _Float16* yh     = (_Float16*)(ws);                     //  8 MB (alias of xh)
  _Float16* wqkvT  = (_Float16*)(ws + (8u  << 20));       //  6 MB [3072][1024]
  _Float16* wprojT = (_Float16*)(ws + (14u << 20));       //  2 MB [1024][1024]
  _Float16* qh     = (_Float16*)(ws + (16u << 20));       //  8 MB [B,H,T,D]
  _Float16* kh     = (_Float16*)(ws + (24u << 20));       //  8 MB [B,H,T,D]
  _Float16* vT     = (_Float16*)(ws + (32u << 20));       //  8 MB [B,H,D,T]

  // 1) convert x to f16
  {
    int n = M_ROWS * EMBD;  // 4194304
    cvt_f32_to_f16_kernel<<<(n + 255) / 256, 256, 0, stream>>>(x, xh, n);
  }
  // 2) transpose+convert weights so WMMA B-fragments are contiguous loads
  {
    int n = EMBD * N_QKV;
    transpose_cvt_kernel<<<(n + 255) / 256, 256, 0, stream>>>(w_attn, wqkvT, EMBD, N_QKV);
  }
  {
    int n = EMBD * EMBD;
    transpose_cvt_kernel<<<(n + 255) / 256, 256, 0, stream>>>(w_proj, wprojT, EMBD, EMBD);
  }
  // 3) QKV projection (WMMA), scatter into q/k/vT head layouts
  qkv_gemm_kernel<<<(M_ROWS / 32) * (N_QKV / 32), 32, 0, stream>>>(
      xh, wqkvT, b_attn, qh, kh, vT);
  // 4) causal flash attention (WMMA for QK^T and PV)
  flash_attn_kernel<<<BATCH * NHEAD * (SEQ / 16), 32, 0, stream>>>(qh, kh, vT, yh);
  // 5) output projection (WMMA) -> fp32 out + bias
  proj_gemm_kernel<<<(M_ROWS / 32) * (EMBD / 32), 32, 0, stream>>>(
      yh, wprojT, b_proj, out);
}
